// ClassifierMutagenicity_50182397886862
// MI455X (gfx1250) — matrix operand order repaired and checked by reading, not compiled, not run on hardware
//
#include <hip/hip_runtime.h>

#define N_NODES   100000
#define N_EDGES   300000
#define N_GRAPHS  2048
#define DIM       256
#define NF        14

typedef __attribute__((ext_vector_type(8)))  float          v8f;
typedef __attribute__((ext_vector_type(16))) __bf16         v16bf;
typedef __attribute__((ext_vector_type(16))) unsigned short v16u;
typedef __attribute__((ext_vector_type(8)))  unsigned short u16x8;

__device__ __forceinline__ unsigned short f2bf(float f) {
    unsigned u = __builtin_bit_cast(unsigned, f);
    u += 0x7FFFu + ((u >> 16) & 1u);          // round-to-nearest-even
    return (unsigned short)(u >> 16);
}
__device__ __forceinline__ float bf2f(unsigned short u) {
    unsigned v = ((unsigned)u) << 16;
    return __builtin_bit_cast(float, v);
}
__device__ __forceinline__ void atomic_add_f32(float* p, float v) {
    __hip_atomic_fetch_add(p, v, __ATOMIC_RELAXED, __HIP_MEMORY_SCOPE_AGENT);
}

// ---------------------------------------------------------------- utilities
__global__ void k_zero4(float* __restrict__ p, int n4) {
    int i = blockIdx.x * blockDim.x + threadIdx.x;
    if (i < n4) ((float4*)p)[i] = make_float4(0.f, 0.f, 0.f, 0.f);
}

// WT[n*Kpad + k] = bf16(W[k*N + n]), zero pad for k >= K
__global__ void k_prep_w(const float* __restrict__ W, unsigned short* __restrict__ WT,
                         int K, int N, int Kpad) {
    int t = blockIdx.x * blockDim.x + threadIdx.x;
    if (t >= N * Kpad) return;
    int n = t / Kpad, k = t - n * Kpad;
    WT[(size_t)n * Kpad + k] = (k < K) ? f2bf(W[(size_t)k * N + n]) : (unsigned short)0;
}

// x [N_NODES][14] f32 -> xpad [N_NODES][32] bf16 (zero padded)
__global__ void k_xpad(const float* __restrict__ x, unsigned short* __restrict__ xp) {
    int t = blockIdx.x * blockDim.x + threadIdx.x;
    if (t >= N_NODES * 32) return;
    int node = t >> 5, k = t & 31;
    xp[t] = (k < NF) ? f2bf(x[(size_t)node * NF + k]) : (unsigned short)0;
}

// ---------------------------------------------------------------- scatter-add
__global__ void k_scatter_l1(const float* __restrict__ x, const int* __restrict__ ei,
                             float* __restrict__ agg) {
    long long t = (long long)blockIdx.x * blockDim.x + threadIdx.x;
    if (t >= (long long)N_EDGES * NF) return;
    int e = (int)(t / NF), k = (int)(t - (long long)e * NF);
    int s = ei[e], d = ei[N_EDGES + e];
    atomic_add_f32(&agg[(size_t)d * 32 + k], x[(size_t)s * NF + k]);
}

__global__ void k_scatter_bf16(const unsigned short* __restrict__ h, const int* __restrict__ ei,
                               float* __restrict__ agg) {
    unsigned t = blockIdx.x * blockDim.x + threadIdx.x;
    if (t >= (unsigned)N_EDGES * 128u) return;
    int e = t >> 7, kp = (t & 127) << 1;
    int s = ei[e], d = ei[N_EDGES + e];
    unsigned pk = *(const unsigned*)(h + (size_t)s * DIM + kp);
    atomic_add_f32(&agg[(size_t)d * DIM + kp + 0], bf2f((unsigned short)(pk & 0xFFFFu)));
    atomic_add_f32(&agg[(size_t)d * DIM + kp + 1], bf2f((unsigned short)(pk >> 16)));
}

__global__ void k_pool(const unsigned short* __restrict__ h, const int* __restrict__ batch,
                       float* __restrict__ g) {
    unsigned t = blockIdx.x * blockDim.x + threadIdx.x;
    if (t >= (unsigned)N_NODES * 128u) return;
    int i = t >> 7, kp = (t & 127) << 1;
    int b = batch[i];
    unsigned pk = *(const unsigned*)(h + (size_t)i * DIM + kp);
    atomic_add_f32(&g[(size_t)b * DIM + kp + 0], bf2f((unsigned short)(pk & 0xFFFFu)));
    atomic_add_f32(&g[(size_t)b * DIM + kp + 1], bf2f((unsigned short)(pk >> 16)));
}

// ---------------------------------------------------------------- WMMA fragment loads
// A fragment (16x32 bf16): lane = {col = M row within tile (lane&15), half = lane>>4}
// per lane: K chunk [kbase, kbase+16), kbase = slice*32 + half*16
__device__ __forceinline__ v16bf load_a_f32(const float* __restrict__ rowptr, int kbase) {
    const float4* p0 = (const float4*)(rowptr + kbase);
    const float4* p1 = (const float4*)(rowptr + kbase + 8);
    float4 f0 = p0[0], f1 = p0[1], f2 = p1[0], f3 = p1[1];
    v16u a;
    a[0] = f2bf(f0.x); a[1] = f2bf(f0.y); a[2]  = f2bf(f0.z); a[3]  = f2bf(f0.w);
    a[4] = f2bf(f1.x); a[5] = f2bf(f1.y); a[6]  = f2bf(f1.z); a[7]  = f2bf(f1.w);
    a[8] = f2bf(f2.x); a[9] = f2bf(f2.y); a[10] = f2bf(f2.z); a[11] = f2bf(f2.w);
    a[12]= f2bf(f3.x); a[13]= f2bf(f3.y); a[14] = f2bf(f3.z); a[15] = f2bf(f3.w);
    return __builtin_bit_cast(v16bf, a);
}
__device__ __forceinline__ v16bf load_a_bf16(const unsigned short* __restrict__ rowptr, int kbase) {
    u16x8 lo = *(const u16x8*)(rowptr + kbase);
    u16x8 hi = *(const u16x8*)(rowptr + kbase + 8);
    v16u a;
#pragma unroll
    for (int i = 0; i < 8; ++i) { a[i] = lo[i]; a[8 + i] = hi[i]; }
    return __builtin_bit_cast(v16bf, a);
}

// 16 N-tiles of WMMA against one A fragment, with double-buffered B loads
// (batches of 4 fragments in flight -> partial loadcnt waits instead of full drains)
__device__ __forceinline__ void wmma_row(v8f (&acc)[16], v16bf av,
                                         const unsigned short* __restrict__ WT,
                                         int K, int kb, int col) {
    v16u b0[4], b1[4];
#pragma unroll
    for (int j = 0; j < 4; ++j)
        b0[j] = *(const v16u*)(WT + (size_t)((j << 4) + col) * K + kb);
#pragma unroll
    for (int tb = 0; tb < 16; tb += 4) {
        if (tb < 12) {
#pragma unroll
            for (int j = 0; j < 4; ++j)
                b1[j] = *(const v16u*)(WT + (size_t)(((tb + 4 + j) << 4) + col) * K + kb);
        }
#pragma unroll
        for (int j = 0; j < 4; ++j)
            acc[tb + j] = __builtin_amdgcn_wmma_f32_16x16x32_bf16(
                false, av, false, __builtin_bit_cast(v16bf, b0[j]), (short)0,
                acc[tb + j], false, false);
        if (tb < 12) {
#pragma unroll
            for (int j = 0; j < 4; ++j) b0[j] = b1[j];
        }
    }
}

// ---------------------------------------------------------------- GraphConv GEMM
// out[m][n] = relu( sum_k agg[m][k] WrelT[n][k] + sum_k h[m][k] WrootT[n][k] + bias[n] )
template <int SA, int SB>
__global__ __launch_bounds__(256) void k_gemm_conv(
    const float* __restrict__ agg,
    const unsigned short* __restrict__ hp,
    const unsigned short* __restrict__ WrelT,
    const unsigned short* __restrict__ WrootT,
    const float* __restrict__ bias,
    unsigned short* __restrict__ hout, int M)
{
    int wave = blockIdx.x * (blockDim.x >> 5) + (threadIdx.x >> 5);
    int m0 = wave << 4;
    if (m0 >= M) return;                       // wave-uniform exit: EXEC stays all-ones
    int lane = threadIdx.x & 31;
    int col  = lane & 15;                      // M-row (A) / N-col (B,C/D) within tile
    int half = lane >> 4;
    const int Ka = SA << 5, Kb = SB << 5;

    v8f acc[16];
#pragma unroll
    for (int t = 0; t < 16; ++t) acc[t] = (v8f){0.f,0.f,0.f,0.f,0.f,0.f,0.f,0.f};

    // ---- part 1: agg (f32 -> bf16) x WrelT
    const float* arow = agg + (size_t)(m0 + col) * Ka;
#pragma unroll
    for (int s = 0; s < SA; ++s) {
        v16bf av = load_a_f32(arow, (s << 5) + (half << 4));
        wmma_row(acc, av, WrelT, Ka, (s << 5) + (half << 4), col);
    }
    // ---- part 2: h (bf16) x WrootT
    const unsigned short* hrow = hp + (size_t)(m0 + col) * Kb;
#pragma unroll
    for (int s = 0; s < SB; ++s) {
        v16bf av = load_a_bf16(hrow, (s << 5) + (half << 4));
        wmma_row(acc, av, WrootT, Kb, (s << 5) + (half << 4), col);
    }
    // ---- epilogue: bias + relu -> bf16
#pragma unroll
    for (int t = 0; t < 16; ++t) {
        int n = (t << 4) + col;
        float bv = bias[n];
#pragma unroll
        for (int v = 0; v < 8; ++v) {
            float val = acc[t][v] + bv;
            val = val > 0.f ? val : 0.f;
            hout[(size_t)(m0 + (half << 3) + v) * DIM + n] = f2bf(val);
        }
    }
}

// ---------------------------------------------------------------- lin1 GEMM (f32 in/out)
template <int S>
__global__ __launch_bounds__(256) void k_gemm_lin(
    const float* __restrict__ A,
    const unsigned short* __restrict__ WT,
    const float* __restrict__ bias,
    float* __restrict__ out, int M)
{
    int wave = blockIdx.x * (blockDim.x >> 5) + (threadIdx.x >> 5);
    int m0 = wave << 4;
    if (m0 >= M) return;
    int lane = threadIdx.x & 31;
    int col = lane & 15, half = lane >> 4;
    const int K = S << 5;

    v8f acc[16];
#pragma unroll
    for (int t = 0; t < 16; ++t) acc[t] = (v8f){0.f,0.f,0.f,0.f,0.f,0.f,0.f,0.f};

    const float* arow = A + (size_t)(m0 + col) * K;
#pragma unroll
    for (int s = 0; s < S; ++s) {
        v16bf av = load_a_f32(arow, (s << 5) + (half << 4));
        wmma_row(acc, av, WT, K, (s << 5) + (half << 4), col);
    }
#pragma unroll
    for (int t = 0; t < 16; ++t) {
        int n = (t << 4) + col;
        float bv = bias[n];
#pragma unroll
        for (int v = 0; v < 8; ++v) {
            float val = acc[t][v] + bv;
            out[(size_t)(m0 + (half << 3) + v) * DIM + n] = val > 0.f ? val : 0.f;
        }
    }
}

// ---------------------------------------------------------------- head: lin2 + log_softmax
__global__ void k_head(const float* __restrict__ g2, const float* __restrict__ w,
                       const float* __restrict__ b, float* __restrict__ out) {
    int gid = blockIdx.x * blockDim.x + threadIdx.x;
    if (gid >= N_GRAPHS) return;
    const float* row = g2 + (size_t)gid * DIM;
    float z0 = b[0], z1 = b[1];
    for (int k = 0; k < DIM; ++k) {
        float v = row[k];
        z0 += v * w[k * 2 + 0];
        z1 += v * w[k * 2 + 1];
    }
    float mx = fmaxf(z0, z1);
    float lse = mx + __logf(__expf(z0 - mx) + __expf(z1 - mx));
    out[gid * 2 + 0] = z0 - lse;
    out[gid * 2 + 1] = z1 - lse;
}

// ---------------------------------------------------------------- launch
extern "C" void kernel_launch(void* const* d_in, const int* in_sizes, int n_in,
                              void* d_out, int out_size, void* d_ws, size_t ws_size,
                              hipStream_t stream) {
    const float* x       = (const float*)d_in[0];
    const int*   ei      = (const int*)d_in[1];
    const int*   batch   = (const int*)d_in[2];
    const float* rel[5]  = {(const float*)d_in[3], (const float*)d_in[6], (const float*)d_in[9],
                            (const float*)d_in[12], (const float*)d_in[15]};
    const float* relb[5] = {(const float*)d_in[4], (const float*)d_in[7], (const float*)d_in[10],
                            (const float*)d_in[13], (const float*)d_in[16]};
    const float* root[5] = {(const float*)d_in[5], (const float*)d_in[8], (const float*)d_in[11],
                            (const float*)d_in[14], (const float*)d_in[17]};
    const float* lin1_w = (const float*)d_in[18];
    const float* lin1_b = (const float*)d_in[19];
    const float* lin2_w = (const float*)d_in[20];
    const float* lin2_b = (const float*)d_in[21];
    float* out = (float*)d_out;

    // ---- workspace carve (256B aligned)
    char* p = (char*)d_ws;
    auto carve = [&](size_t bytes) -> void* {
        void* r = (void*)p;
        p += (bytes + 255) & ~(size_t)255;
        return r;
    };
    float*          agg   = (float*)carve((size_t)N_NODES * DIM * 4);
    unsigned short* hA    = (unsigned short*)carve((size_t)N_NODES * DIM * 2);
    unsigned short* hB    = (unsigned short*)carve((size_t)N_NODES * DIM * 2);
    unsigned short* xpad  = (unsigned short*)carve((size_t)N_NODES * 32 * 2);
    unsigned short* WrelT[5];
    unsigned short* WrootT[5];
    for (int i = 0; i < 5; ++i) {
        WrelT[i]  = (unsigned short*)carve((size_t)DIM * DIM * 2);
        WrootT[i] = (unsigned short*)carve((size_t)DIM * DIM * 2);
    }
    unsigned short* lin1T = (unsigned short*)carve((size_t)DIM * DIM * 2);
    float*          gpool = (float*)carve((size_t)N_GRAPHS * DIM * 4);
    float*          g2    = (float*)carve((size_t)N_GRAPHS * DIM * 4);

    // ---- weight prep
    k_prep_w<<<(DIM * 32 + 255) / 256, 256, 0, stream>>>(rel[0],  WrelT[0],  NF, DIM, 32);
    k_prep_w<<<(DIM * 32 + 255) / 256, 256, 0, stream>>>(root[0], WrootT[0], NF, DIM, 32);
    for (int i = 1; i < 5; ++i) {
        k_prep_w<<<(DIM * DIM) / 256, 256, 0, stream>>>(rel[i],  WrelT[i],  DIM, DIM, DIM);
        k_prep_w<<<(DIM * DIM) / 256, 256, 0, stream>>>(root[i], WrootT[i], DIM, DIM, DIM);
    }
    k_prep_w<<<(DIM * DIM) / 256, 256, 0, stream>>>(lin1_w, lin1T, DIM, DIM, DIM);
    k_xpad<<<(N_NODES * 32 + 255) / 256, 256, 0, stream>>>(x, xpad);

    const int gemmGrid = ((N_NODES / 16) + 7) / 8;   // 8 waves/block

    // ---- layer 1 (K padded 14 -> 32)
    k_zero4<<<(N_NODES * 32 / 4 + 255) / 256, 256, 0, stream>>>(agg, N_NODES * 32 / 4);
    k_scatter_l1<<<(int)(((long long)N_EDGES * NF + 255) / 256), 256, 0, stream>>>(x, ei, agg);
    k_gemm_conv<1, 1><<<gemmGrid, 256, 0, stream>>>(agg, xpad, WrelT[0], WrootT[0], relb[0], hA, N_NODES);

    // ---- layers 2..5
    unsigned short* hin = hA;
    unsigned short* hout = hB;
    for (int L = 1; L < 5; ++L) {
        k_zero4<<<(N_NODES * DIM / 4 + 255) / 256, 256, 0, stream>>>(agg, N_NODES * DIM / 4);
        k_scatter_bf16<<<(N_EDGES * 128 + 255) / 256, 256, 0, stream>>>(hin, ei, agg);
        k_gemm_conv<8, 8><<<gemmGrid, 256, 0, stream>>>(agg, hin, WrelT[L], WrootT[L], relb[L], hout, N_NODES);
        unsigned short* t = hin; hin = hout; hout = t;
    }

    // ---- global add pool
    k_zero4<<<(N_GRAPHS * DIM / 4 + 255) / 256, 256, 0, stream>>>(gpool, N_GRAPHS * DIM / 4);
    k_pool<<<(N_NODES * 128 + 255) / 256, 256, 0, stream>>>(hin, batch, gpool);

    // ---- lin1 + relu (M = 2048 -> 128 waves -> 16 blocks)
    k_gemm_lin<8><<<((N_GRAPHS / 16) + 7) / 8, 256, 0, stream>>>(gpool, lin1T, lin1_b, g2, N_GRAPHS);

    // ---- lin2 + log_softmax
    k_head<<<(N_GRAPHS + 255) / 256, 256, 0, stream>>>(g2, lin2_w, lin2_b, out);
}
